// Block_2095944040561
// MI455X (gfx1250) — matrix operand order repaired and checked by reading, not compiled, not run on hardware
//
#include <hip/hip_runtime.h>
#include <hip/hip_bf16.h>
#include <math.h>

// ---------------------------------------------------------------------------
// Types / WMMA helpers (CDNA5 gfx1250, wave32)
// ---------------------------------------------------------------------------
typedef _Float16 f16;
typedef __attribute__((ext_vector_type(16))) _Float16 v16h;
typedef __attribute__((ext_vector_type(8)))  _Float16 v8h;
typedef __attribute__((ext_vector_type(4)))  _Float16 v4h;
typedef __attribute__((ext_vector_type(8)))  float    v8f;

#define WMMA16(a, b, c) \
  __builtin_amdgcn_wmma_f32_16x16x32_f16(false, (a), false, (b), (short)0, (c), false, false)

union AF { v16h v; v8h h[2]; };

__device__ __forceinline__ v8f zero8() {
  v8f z;
#pragma unroll
  for (int i = 0; i < 8; i++) z[i] = 0.f;
  return z;
}

// A-matrix 16x32 f16 fragment (ISA 7.12.2): row = lane&15,
// halves[0..7] = K ((lane>=16)?8:0)+0..7, halves[8..15] = same +16.
__device__ __forceinline__ v16h load_A16(const f16* __restrict__ p, int ld) {
  int lane = threadIdx.x & 31;
  const f16* r = p + (size_t)(lane & 15) * ld + ((lane >> 4) << 3);
  AF f;
  f.h[0] = *(const v8h*)r;
  f.h[1] = *(const v8h*)(r + 16);
  return f.v;
}

// B-matrix 32x16 f16 fragment: column N = lane&15 (a row of the source
// matrix stored K-contiguous), halves[0..7] = K ((lane>=16)?8:0)+0..7, halves[8..15] = same +16.
__device__ __forceinline__ v16h load_B16(const f16* __restrict__ p, int ld) {
  int lane = threadIdx.x & 31;
  const f16* r = p + (size_t)(lane & 15) * ld + ((lane >> 4) << 3);
  AF f;
  f.h[0] = *(const v8h*)r;
  f.h[1] = *(const v8h*)(r + 16);
  return f.v;
}

// C/D frag (f32 16x16): element e -> row = e + 8*(lane>>4), col = lane&15.

// ---------------------------------------------------------------------------
// Shared GEMM core: one wave computes a 16x64 tile of  C = A(NxK) * W(MxK)^T
// ---------------------------------------------------------------------------
__device__ __forceinline__ void gemm16x64(const f16* __restrict__ A,
                                          const f16* __restrict__ W,
                                          int row0, int col0, int Kd, v8f c[4]) {
  for (int k = 0; k < Kd; k += 32) {
    v16h a = load_A16(A + (size_t)row0 * Kd + k, Kd);
#pragma unroll
    for (int j = 0; j < 4; j++) {
      v16h b = load_B16(W + (size_t)(col0 + 16 * j) * Kd + k, Kd);
      c[j] = WMMA16(a, b, c[j]);
    }
  }
}

// ---------------------------------------------------------------------------
// fp32 -> f16 weight conversion
// ---------------------------------------------------------------------------
__global__ void cvt_f16_kernel(const float* __restrict__ in, f16* __restrict__ out, int n) {
  int i = blockIdx.x * 256 + threadIdx.x;
  if (i < n) out[i] = (f16)in[i];
}

// ---------------------------------------------------------------------------
// LayerNorm (fp32 math) -> f16 output. One wave per 512-elem row.
// ---------------------------------------------------------------------------
__global__ void ln_f16_kernel(const float* __restrict__ x, const float* __restrict__ w,
                              const float* __restrict__ b, f16* __restrict__ out) {
  int row  = blockIdx.x * 8 + (threadIdx.x >> 5);
  int lane = threadIdx.x & 31;
  const float* xr = x + (size_t)row * 512 + lane * 16;
  float v[16];
#pragma unroll
  for (int i = 0; i < 16; i += 4) *(float4*)(v + i) = *(const float4*)(xr + i);
  float s = 0.f, ss = 0.f;
#pragma unroll
  for (int i = 0; i < 16; i++) { s += v[i]; ss += v[i] * v[i]; }
#pragma unroll
  for (int m = 1; m <= 16; m <<= 1) { s += __shfl_xor(s, m, 32); ss += __shfl_xor(ss, m, 32); }
  float mean = s * (1.f / 512.f);
  float var  = ss * (1.f / 512.f) - mean * mean;
  float r    = rsqrtf(var + 1e-5f);
  v16h o;
#pragma unroll
  for (int i = 0; i < 16; i++)
    o[i] = (f16)((v[i] - mean) * r * w[lane * 16 + i] + b[lane * 16 + i]);
  *(v16h*)(out + (size_t)row * 512 + lane * 16) = o;
}

// ---------------------------------------------------------------------------
// QKV GEMM: h[N,512] * Wqkv[1536,512]^T + bias, scatter to
//   Q,K : [BH=32, S=2048, hd=128] row-major (f16)
//   VT  : [BH=32, hd=128, S=2048]  (V transposed; makes P*V B-frags contiguous)
// ---------------------------------------------------------------------------
__global__ void gemm_qkv_kernel(const f16* __restrict__ A, const f16* __restrict__ W,
                                const float* __restrict__ bias,
                                f16* __restrict__ Q, f16* __restrict__ K,
                                f16* __restrict__ VT) {
  int wv = threadIdx.x >> 5, lane = threadIdx.x & 31;
  int row0 = blockIdx.x * 128 + wv * 16;
  int col0 = blockIdx.y * 64;
  v8f c[4];
#pragma unroll
  for (int j = 0; j < 4; j++) c[j] = zero8();
  gemm16x64(A, W, row0, col0, 512, c);

  int hi = lane >> 4, nn = lane & 15;
#pragma unroll
  for (int j = 0; j < 4; j++) {
    int m = col0 + 16 * j + nn;
    int part = m >> 9;           // 0=q 1=k 2=v
    int f = m & 511;
    int hh = f >> 7, d = f & 127;
    float bv = bias[m];
#pragma unroll
    for (int e = 0; e < 8; e++) {
      int t = row0 + e + 8 * hi;           // token
      int bidx = t >> 11, s = t & 2047;
      f16 val = (f16)(c[j][e] + bv);
      size_t bh = (size_t)(bidx * 4 + hh);
      if (part == 0)      Q [(bh * 2048 + s) * 128 + d] = val;
      else if (part == 1) K [(bh * 2048 + s) * 128 + d] = val;
      else                VT[(bh * 128 + d) * 2048 + s] = val;
    }
  }
}

// ---------------------------------------------------------------------------
// GEMM + bias + fp32 residual -> fp32 out[N,512]. (out may alias res.)
// ---------------------------------------------------------------------------
__global__ void gemm_bias_res_kernel(const f16* __restrict__ A, const f16* __restrict__ W,
                                     const float* __restrict__ bias,
                                     const float* res, float* out, int Kd) {
  int wv = threadIdx.x >> 5, lane = threadIdx.x & 31;
  int row0 = blockIdx.x * 128 + wv * 16;
  int col0 = blockIdx.y * 64;
  v8f c[4];
#pragma unroll
  for (int j = 0; j < 4; j++) c[j] = zero8();
  gemm16x64(A, W, row0, col0, Kd, c);

  int hi = lane >> 4, nn = lane & 15;
#pragma unroll
  for (int j = 0; j < 4; j++) {
    int col = col0 + 16 * j + nn;
    float bv = bias[col];
#pragma unroll
    for (int e = 0; e < 8; e++) {
      int t = row0 + e + 8 * hi;
      size_t idx = (size_t)t * 512 + col;
      out[idx] = c[j][e] + bv + res[idx];
    }
  }
}

// ---------------------------------------------------------------------------
// GEMM + bias + exact-erf GELU -> f16 out[N,1024]
// ---------------------------------------------------------------------------
__global__ void gemm_gelu_kernel(const f16* __restrict__ A, const f16* __restrict__ W,
                                 const float* __restrict__ bias, f16* __restrict__ out) {
  int wv = threadIdx.x >> 5, lane = threadIdx.x & 31;
  int row0 = blockIdx.x * 128 + wv * 16;
  int col0 = blockIdx.y * 64;
  v8f c[4];
#pragma unroll
  for (int j = 0; j < 4; j++) c[j] = zero8();
  gemm16x64(A, W, row0, col0, 512, c);

  int hi = lane >> 4, nn = lane & 15;
#pragma unroll
  for (int j = 0; j < 4; j++) {
    int col = col0 + 16 * j + nn;
    float bv = bias[col];
#pragma unroll
    for (int e = 0; e < 8; e++) {
      int t = row0 + e + 8 * hi;
      float v = c[j][e] + bv;
      float g = 0.5f * v * (1.0f + erff(v * 0.70710678118654752f));
      out[(size_t)t * 1024 + col] = (f16)g;
    }
  }
}

// ---------------------------------------------------------------------------
// Local banded attention (|i-j| <= 5, 11 keys): tiny FLOPs -> VALU kernel.
// One wave per (bh, s) row; lane owns 4 of the 128 head dims.
// ---------------------------------------------------------------------------
__global__ void local_attn_kernel(const f16* __restrict__ Q, const f16* __restrict__ K,
                                  const f16* __restrict__ VT, f16* __restrict__ AO) {
  int g = blockIdx.x * 8 + (threadIdx.x >> 5);
  int lane = threadIdx.x & 31;
  int bh = g >> 11, s = g & 2047;
  int d0 = lane * 4;

  v4h qv = *(const v4h*)(Q + ((size_t)bh * 2048 + s) * 128 + d0);
  float qf[4];
#pragma unroll
  for (int i = 0; i < 4; i++) qf[i] = (float)qv[i];

  float sc[11];
#pragma unroll
  for (int jj = 0; jj < 11; jj++) {
    int j = s - 5 + jj;
    float p = -1e30f;
    if (j >= 0 && j < 2048) {                       // uniform per wave
      v4h kv = *(const v4h*)(K + ((size_t)bh * 2048 + j) * 128 + d0);
      p = qf[0] * (float)kv[0] + qf[1] * (float)kv[1] +
          qf[2] * (float)kv[2] + qf[3] * (float)kv[3];
#pragma unroll
      for (int m = 1; m <= 16; m <<= 1) p += __shfl_xor(p, m, 32);
      p *= 0.08838834764831845f;                     // 1/sqrt(128)
    }
    sc[jj] = p;
  }
  float mx = -1e30f;
#pragma unroll
  for (int jj = 0; jj < 11; jj++) mx = fmaxf(mx, sc[jj]);
  float l = 0.f, pr[11];
#pragma unroll
  for (int jj = 0; jj < 11; jj++) {
    int j = s - 5 + jj;
    float e = (j >= 0 && j < 2048) ? __expf(sc[jj] - mx) : 0.f;
    pr[jj] = e; l += e;
  }
  float inv = 1.f / l;
  float o[4] = {0.f, 0.f, 0.f, 0.f};
#pragma unroll
  for (int jj = 0; jj < 11; jj++) {
    int j = s - 5 + jj;
    if (j >= 0 && j < 2048) {
      const f16* vr = VT + ((size_t)bh * 128 + d0) * 2048 + j;
#pragma unroll
      for (int i = 0; i < 4; i++) o[i] += pr[jj] * (float)vr[(size_t)i * 2048];
    }
  }
  int b_ = bh >> 2, h = bh & 3;
  v4h ov;
#pragma unroll
  for (int i = 0; i < 4; i++) ov[i] = (f16)(o[i] * inv);
  *(v4h*)(AO + ((size_t)(b_ * 2048 + s)) * 512 + h * 128 + d0) = ov;
}

// ---------------------------------------------------------------------------
// Global attention, flash-style. Block = 8 waves; wave owns 16 query rows,
// streams all 2048 keys in 32-key steps: 8 WMMA (Q*K^T) + online softmax
// + P staged via LDS (C-frag -> A-frag relayout) + 8 WMMA (P * V^T).
// ---------------------------------------------------------------------------
__global__ void __launch_bounds__(256, 1)
global_attn_kernel(const f16* __restrict__ Q, const f16* __restrict__ K,
                   const f16* __restrict__ VT, f16* __restrict__ AO) {
  __shared__ f16 Pl[8][16 * 40];                       // per-wave 16x32 P tile, 80B row stride
  int wv = threadIdx.x >> 5, lane = threadIdx.x & 31;
  int hi = lane >> 4, nn = lane & 15;
  int bh = blockIdx.y;
  int q0 = blockIdx.x * 128 + wv * 16;

  const f16* Qb = Q  + (size_t)bh * 2048 * 128;
  const f16* Kb = K  + (size_t)bh * 2048 * 128;
  const f16* Vb = VT + (size_t)bh * 128 * 2048;

  v16h aq[4];
#pragma unroll
  for (int fc = 0; fc < 4; fc++) aq[fc] = load_A16(Qb + (size_t)q0 * 128 + fc * 32, 128);

  v8f O[8];
  float m8[8], l8[8];
#pragma unroll
  for (int e = 0; e < 8; e++) { m8[e] = -1e30f; l8[e] = 0.f; }
#pragma unroll
  for (int dc = 0; dc < 8; dc++) O[dc] = zero8();

  f16* myP = &Pl[wv][0];
  const float scale = 0.08838834764831845f;

  for (int k0 = 0; k0 < 2048; k0 += 32) {
    v8f s0 = zero8(), s1 = zero8();
#pragma unroll
    for (int fc = 0; fc < 4; fc++) {
      v16h bk0 = load_B16(Kb + (size_t)k0 * 128 + fc * 32, 128);
      v16h bk1 = load_B16(Kb + (size_t)(k0 + 16) * 128 + fc * 32, 128);
      s0 = WMMA16(aq[fc], bk0, s0);
      s1 = WMMA16(aq[fc], bk1, s1);
    }
    float corr[8];
#pragma unroll
    for (int e = 0; e < 8; e++) {
      float a0 = s0[e] * scale, a1 = s1[e] * scale;
      float mx = fmaxf(a0, a1);
#pragma unroll
      for (int m = 1; m <= 8; m <<= 1) mx = fmaxf(mx, __shfl_xor(mx, m, 32));
      float mn = fmaxf(m8[e], mx);
      corr[e] = __expf(m8[e] - mn);
      m8[e] = mn;
      float p0 = __expf(a0 - mn), p1 = __expf(a1 - mn);
      float rs = p0 + p1;
#pragma unroll
      for (int m = 1; m <= 8; m <<= 1) rs += __shfl_xor(rs, m, 32);
      l8[e] = l8[e] * corr[e] + rs;
      int row = e + 8 * hi;
      myP[row * 40 + nn]      = (f16)p0;    // keys k0..k0+15
      myP[row * 40 + 16 + nn] = (f16)p1;    // keys k0+16..k0+31
    }
#pragma unroll
    for (int dc = 0; dc < 8; dc++)
#pragma unroll
      for (int e = 0; e < 8; e++) O[dc][e] *= corr[e];

    __syncthreads();
    v16h ap;
    {
      int ko = hi * 8;
      AF f;
      f.h[0] = *(const v8h*)&myP[nn * 40 + ko];
      f.h[1] = *(const v8h*)&myP[nn * 40 + 16 + ko];
      ap = f.v;
    }
#pragma unroll
    for (int dc = 0; dc < 8; dc++) {
      v16h bv = load_B16(Vb + (size_t)(dc * 16) * 2048 + k0, 2048);
      O[dc] = WMMA16(ap, bv, O[dc]);
    }
    __syncthreads();
  }

  int b_ = bh >> 2, h = bh & 3;
#pragma unroll
  for (int e = 0; e < 8; e++) {
    int sq = q0 + e + 8 * hi;
    float inv = 1.f / l8[e];
    f16* ao = AO + ((size_t)(b_ * 2048 + sq)) * 512 + h * 128 + nn;
#pragma unroll
    for (int dc = 0; dc < 8; dc++) ao[dc * 16] = (f16)(O[dc][e] * inv);
  }
}

// ---------------------------------------------------------------------------
// Host orchestration
// ---------------------------------------------------------------------------
extern "C" void kernel_launch(void* const* d_in, const int* in_sizes, int n_in,
                              void* d_out, int out_size, void* d_ws, size_t ws_size,
                              hipStream_t stream) {
  (void)in_sizes; (void)n_in; (void)out_size; (void)ws_size;

  const float* x      = (const float*)d_in[0];
  const float* ln1w   = (const float*)d_in[1];
  const float* ln1b   = (const float*)d_in[2];
  const float* ln2w   = (const float*)d_in[3];
  const float* ln2b   = (const float*)d_in[4];
  const float* ln3w   = (const float*)d_in[5];
  const float* ln3b   = (const float*)d_in[6];
  const float* Wqkv_l = (const float*)d_in[7];
  const float* bqkv_l = (const float*)d_in[8];
  const float* Wo_l   = (const float*)d_in[9];
  const float* bo_l   = (const float*)d_in[10];
  const float* Wqkv_g = (const float*)d_in[11];
  const float* bqkv_g = (const float*)d_in[12];
  const float* Wo_g   = (const float*)d_in[13];
  const float* bo_g   = (const float*)d_in[14];
  const float* W1     = (const float*)d_in[15];
  const float* b1     = (const float*)d_in[16];
  const float* W2     = (const float*)d_in[17];
  const float* b2     = (const float*)d_in[18];

  char* ws = (char*)d_ws;
  f16*   wqkv_l16 = (f16*)(ws + 0);          // 1536*512 f16
  f16*   wo_l16   = (f16*)(ws + 1572864);    // 512*512
  f16*   wqkv_g16 = (f16*)(ws + 2097152);    // 1536*512
  f16*   wo_g16   = (f16*)(ws + 3670016);    // 512*512
  f16*   w1_16    = (f16*)(ws + 4194304);    // 1024*512
  f16*   w2_16    = (f16*)(ws + 5242880);    // 512*1024
  f16*   h16      = (f16*)(ws + 6291456);    // [16384,512]  f16
  f16*   Qb       = (f16*)(ws + 23068672);   // [32,2048,128] f16
  f16*   Kb       = (f16*)(ws + 39845888);   // [32,2048,128] f16
  f16*   VTb      = (f16*)(ws + 56623104);   // [32,128,2048] f16
  f16*   AO       = (f16*)(ws + 73400320);   // [16384,512]  f16
  float* x2       = (float*)(ws + 90177536); // [16384,512]  f32
  f16*   tbuf     = (f16*)(ws + 23068672);   // [16384,1024] f16, aliases Q/K (attention done)

  // weight conversion fp32 -> f16
  cvt_f16_kernel<<<3072, 256, 0, stream>>>(Wqkv_l, wqkv_l16, 786432);
  cvt_f16_kernel<<<1024, 256, 0, stream>>>(Wo_l,   wo_l16,   262144);
  cvt_f16_kernel<<<3072, 256, 0, stream>>>(Wqkv_g, wqkv_g16, 786432);
  cvt_f16_kernel<<<1024, 256, 0, stream>>>(Wo_g,   wo_g16,   262144);
  cvt_f16_kernel<<<2048, 256, 0, stream>>>(W1,     w1_16,    524288);
  cvt_f16_kernel<<<2048, 256, 0, stream>>>(W2,     w2_16,    524288);

  dim3 blk(256);
  dim3 gQKV(128, 24);   // 16384/128 x 1536/64
  dim3 gOUT(128, 8);    // x 512/64
  dim3 gFF1(128, 16);   // x 1024/64

  // ---- Layer 1: local banded attention ----
  ln_f16_kernel<<<2048, blk, 0, stream>>>(x, ln1w, ln1b, h16);
  gemm_qkv_kernel<<<gQKV, blk, 0, stream>>>(h16, wqkv_l16, bqkv_l, Qb, Kb, VTb);
  local_attn_kernel<<<8192, blk, 0, stream>>>(Qb, Kb, VTb, AO);
  gemm_bias_res_kernel<<<gOUT, blk, 0, stream>>>(AO, wo_l16, bo_l, x, x2, 512);

  // ---- Layer 2: global attention ----
  ln_f16_kernel<<<2048, blk, 0, stream>>>(x2, ln2w, ln2b, h16);
  gemm_qkv_kernel<<<gQKV, blk, 0, stream>>>(h16, wqkv_g16, bqkv_g, Qb, Kb, VTb);
  dim3 gATT(16, 32);    // 2048/128 q-tiles x (B*H)
  global_attn_kernel<<<gATT, blk, 0, stream>>>(Qb, Kb, VTb, AO);
  gemm_bias_res_kernel<<<gOUT, blk, 0, stream>>>(AO, wo_g16, bo_g, x2, x2, 512);

  // ---- FFN ----
  ln_f16_kernel<<<2048, blk, 0, stream>>>(x2, ln3w, ln3b, h16);
  gemm_gelu_kernel<<<gFF1, blk, 0, stream>>>(h16, w1_16, b1, tbuf);
  gemm_bias_res_kernel<<<gOUT, blk, 0, stream>>>(tbuf, w2_16, b2, x2, (float*)d_out, 1024);
}
